// MultiHeadAttention_46153718562818
// MI455X (gfx1250) — matrix operand order repaired and checked by reading, not compiled, not run on hardware
//
#include <hip/hip_runtime.h>
#include <cstdint>
#include <cstddef>

// ---------------- types ----------------
typedef __bf16 bf16_t;
typedef __attribute__((ext_vector_type(16))) __bf16 v16bf;
typedef __attribute__((ext_vector_type(8)))  __bf16 bh8;   // 16 bytes
typedef __attribute__((ext_vector_type(4)))  __bf16 bh4;   // 8 bytes
typedef __attribute__((ext_vector_type(8)))  float  v8f;
typedef __attribute__((ext_vector_type(4)))  unsigned int u32x4;
typedef __attribute__((ext_vector_type(8)))  int i32x8;
typedef __attribute__((ext_vector_type(4)))  int i32x4;

static constexpr int Bn = 4, Tn = 2048, Dn = 1024, Hn = 16, HDn = 64;

#define WMMA_BF16(A_, B_, C_) \
  __builtin_amdgcn_wmma_f32_16x16x32_bf16(false, (A_), false, (B_), (short)0, (C_), false, false)

// ---- CDNA5 TDM availability probes ----
#if defined(__has_builtin)
#  if __has_builtin(__builtin_amdgcn_tensor_load_to_lds)
#    define HAVE_TDM 1
#  endif
#  if __has_builtin(__builtin_amdgcn_s_wait_tensorcnt)
#    define HAVE_WAIT_TENSOR 1
#  endif
#endif

static __device__ __forceinline__ void wait_tensorcnt0() {
#if defined(HAVE_WAIT_TENSOR)
  __builtin_amdgcn_s_wait_tensorcnt((short)0);
#else
  asm volatile("s_wait_tensorcnt 0x0" ::: "memory");
#endif
}

static __device__ __forceinline__ v16bf cat8(bh8 lo, bh8 hi) {
  return __builtin_shufflevector(lo, hi, 0,1,2,3,4,5,6,7,8,9,10,11,12,13,14,15);
}

// A fragment (16x32 bf16, M x K) from a row-major tile with row stride 32 elems.
// Lane layout: m = lane&15, half h = lane>>4; per-lane K run [16h, 16h+16).
static __device__ __forceinline__ v16bf frag_a_row32(const bf16_t* base, int m, int lh) {
  const bf16_t* row = base + m * 32;
  bh8 lo = *(const bh8*)(row + 16 * lh);
  bh8 hi = *(const bh8*)(row + 16 * lh + 8);
  return cat8(lo, hi);
}

// B fragment (32x16 bf16, K x N) from a K-major tile: element [n][k], stride 32.
// Lane layout: n = lane&15, half h = lane>>4; per-lane K run [16h, 16h+16).
static __device__ __forceinline__ v16bf frag_b_kmaj32(const bf16_t* base, int n, int lh) {
  const bf16_t* col = base + n * 32 + 16 * lh;
  return cat8(*(const bh8*)col, *(const bh8*)(col + 8));
}

// ---------------- f32 -> bf16 conversion ----------------
__global__ void cvt_f32_bf16(const float* __restrict__ s, bf16_t* __restrict__ d, int n4) {
  int i = blockIdx.x * blockDim.x + threadIdx.x;
  if (i >= n4) return;
  float4 f = ((const float4*)s)[i];
  bh4 o = { (bf16_t)f.x, (bf16_t)f.y, (bf16_t)f.z, (bf16_t)f.w };
  ((bh4*)d)[i] = o;
}

// ---------------- tiled bf16 WMMA GEMM:  C = A(MxK) * W(KxN) + bias ----------------
// Block tile 128x64, BK=32, 8 waves (4x2), each wave 32x32 (2x2 WMMA tiles).
// A tile is staged into LDS by the Tensor Data Mover (TENSOR_LOAD_TO_LDS) when the
// builtin is available (D# built per CDNA5 ISA §8: 2D tile 128x32 bf16, stride K).
// mode 0: store f32 row-major (output projection -> d_out)
// mode 1: store bf16 permuted to (B,H,T,HD) (QKV projections)
__global__ void __launch_bounds__(256)
gemm_bf16_tile(const bf16_t* __restrict__ A, const bf16_t* __restrict__ W,
               const float* __restrict__ bias, float* __restrict__ outF,
               bf16_t* __restrict__ outP, int N, int Kd, int mode) {
  __shared__ __attribute__((aligned(16))) bf16_t As[128 * 32];  // row-major [m][k]
  __shared__ __attribute__((aligned(16))) bf16_t Bt[64 * 32];   // K-major  [n][k]

  const int tid  = threadIdx.x;
  const int lane = tid & 31;
  const int wave = tid >> 5;
  const int nIdx = lane & 15;
  const int lh   = lane >> 4;
  const int wm   = wave & 3;   // 0..3 -> M sub-block
  const int wn   = wave >> 2;  // 0..1 -> N sub-block
  const int rowBase = blockIdx.y * 128;
  const int colBase = blockIdx.x * 64;

  const int bkk  = tid >> 3, bnn  = tid & 7;  // B tile loaders
#if !defined(HAVE_TDM)
  const int arow = tid >> 1, aseg = tid & 1;  // A tile loaders: 2 thr/row
#endif

  v8f acc[2][2] = {};

  for (int k0 = 0; k0 < Kd; k0 += 32) {
    __syncthreads();

#if defined(HAVE_TDM)
    if (wave == 0) {
      // ---- Tensor DMA descriptor (CDNA5 ISA ch.8) for a 2D bf16 tile:
      //      tile_dim = 32(K) x 128(M), tensor_dim = Kd x M, stride = Kd elems.
      const uint64_t ga  = (uint64_t)(uintptr_t)A +
                           (((size_t)rowBase * (size_t)Kd + (size_t)k0) * 2u);
      const uint32_t lds = (uint32_t)(uintptr_t)(&As[0]);  // LDS byte offset
      u32x4 g0 = {};
      g0[0] = 1u;                                       // count=1, user mode
      g0[1] = lds;                                      // lds_addr
      g0[2] = (uint32_t)ga;                             // global_addr[31:0]
      g0[3] = (uint32_t)(ga >> 32) | (2u << 30);        // global_addr[56:32] | type=2
      const uint32_t td0 = (uint32_t)Kd;                // tensor_dim0 (elems)
      const uint32_t td1 = (uint32_t)(Bn * Tn);         // tensor_dim1 = M rows
      i32x8 g1 = {};
      g1[0] = (int)(1u << 16);                          // data_size=1 (2 bytes)
      g1[1] = (int)((td0 & 0xFFFFu) << 16);             // tensor_dim0[15:0]  @bits63:48
      g1[2] = (int)(((td0 >> 16) & 0xFFFFu) |           // tensor_dim0[31:16] @bits79:64
                    ((td1 & 0xFFFFu) << 16));           // tensor_dim1[15:0]  @bits95:80
      g1[3] = (int)(((td1 >> 16) & 0xFFFFu) |           // tensor_dim1[31:16] @bits111:96
                    (32u << 16));                       // tile_dim0=32       @bits127:112
      g1[4] = 128;                                      // tile_dim1=128 (tile_dim2=0)
      g1[5] = (int)td0;                                 // tensor_dim0_stride[31:0]
      g1[6] = 0;                                        // stride0 hi / stride1 lo
      g1[7] = 0;
      i32x4 gz = {};
#  if __clang_major__ >= 23
      i32x8 gz8 = {};
      __builtin_amdgcn_tensor_load_to_lds(g0, g1, gz, gz, gz8, 0);
#  else
      __builtin_amdgcn_tensor_load_to_lds(g0, g1, gz, gz, 0);
#  endif
      wait_tensorcnt0();
    }
#else
    // Fallback: cooperative A tile load 128x32 (row-major, b128 loads/stores)
    {
      const bh8* ga = (const bh8*)(A + (size_t)(rowBase + arow) * Kd + k0 + aseg * 16);
      bh8 aLo = ga[0], aHi = ga[1];
      *(bh8*)(As + arow * 32 + aseg * 16)     = aLo;
      *(bh8*)(As + arow * 32 + aseg * 16 + 8) = aHi;
    }
#endif

    // B tile 32x64 loaded row-major, stored transposed (K-major per column)
    bh8 wrow = *(const bh8*)(W + (size_t)(k0 + bkk) * N + colBase + bnn * 8);
#pragma unroll
    for (int i = 0; i < 8; ++i) Bt[(bnn * 8 + i) * 32 + bkk] = wrow[i];
    __syncthreads();

    v16bf af0 = frag_a_row32(As + (wm * 32) * 32,      nIdx, lh);
    v16bf af1 = frag_a_row32(As + (wm * 32 + 16) * 32, nIdx, lh);
    v16bf bf0 = frag_b_kmaj32(Bt + (wn * 32) * 32,      nIdx, lh);
    v16bf bf1 = frag_b_kmaj32(Bt + (wn * 32 + 16) * 32, nIdx, lh);
    acc[0][0] = WMMA_BF16(af0, bf0, acc[0][0]);
    acc[0][1] = WMMA_BF16(af0, bf1, acc[0][1]);
    acc[1][0] = WMMA_BF16(af1, bf0, acc[1][0]);
    acc[1][1] = WMMA_BF16(af1, bf1, acc[1][1]);
  }

  // epilogue: C layout -> lane holds (M = r + 8*lh, N = nIdx) per VGPR r
#pragma unroll
  for (int i = 0; i < 2; ++i)
#pragma unroll
    for (int j = 0; j < 2; ++j)
#pragma unroll
      for (int r = 0; r < 8; ++r) {
        const int mr = rowBase + wm * 32 + i * 16 + r + 8 * lh;
        const int nc = colBase + wn * 32 + j * 16 + nIdx;
        const float val = acc[i][j][r] + bias[nc];
        if (mode == 0) {
          outF[(size_t)mr * N + nc] = val;
        } else {
          const int b = mr >> 11, t = mr & (Tn - 1);     // M = B*T rows
          const int h = nc >> 6,  hd = nc & (HDn - 1);   // N = H*HD cols
          outP[(((size_t)b * Hn + h) * Tn + t) * HDn + hd] = (bf16_t)val;
        }
      }
}

// ---------------- flash attention (causal, online softmax) ----------------
// Grid: x = T/128 (q blocks), y = B*H. 8 waves; wave w owns q rows [q0, q0+16).
__global__ void __launch_bounds__(256)
flash_attn_bf16(const bf16_t* __restrict__ Q, const bf16_t* __restrict__ K,
                const bf16_t* __restrict__ V, bf16_t* __restrict__ Out) {
  __shared__ __attribute__((aligned(16))) bf16_t Vt[HDn * 32];   // V block transposed: [hd][key]
  __shared__ __attribute__((aligned(16))) bf16_t Ps[8][16 * 32]; // per-wave P scratch [m][key]

  const int tid  = threadIdx.x;
  const int lane = tid & 31;
  const int wave = tid >> 5;
  const int nIdx = lane & 15;
  const int lh   = lane >> 4;

  const int bh   = blockIdx.y;   // b*H + h
  const int bb   = bh >> 4;
  const int hh   = bh & 15;
  const int qBlk = blockIdx.x * 128;
  const int q0   = qBlk + wave * 16;

  const size_t headOff = (size_t)bh * Tn * HDn;
  const bf16_t* Qh = Q + headOff;
  const bf16_t* Kh = K + headOff;
  const bf16_t* Vh = V + headOff;

  // Q fragments: 16x64 = two 16x32 A-fragments, loaded straight from global
  v16bf qf[2];
#pragma unroll
  for (int c = 0; c < 2; ++c) {
    const bf16_t* qp = Qh + (size_t)(q0 + nIdx) * HDn + c * 32;
    qf[c] = cat8(*(const bh8*)(qp + 16 * lh), *(const bh8*)(qp + 16 * lh + 8));
  }

  v8f oAcc[4] = {};            // O tile 16x64 f32 (4 x 16-wide)
  float mRow[8], lRow[8];
#pragma unroll
  for (int r = 0; r < 8; ++r) { mRow[r] = -3.0e38f; lRow[r] = 0.0f; }

  const int qLast = q0 + 15;
  const int kvEnd = qBlk + 128;  // causal: keys beyond WG's last q row are dead

  for (int kv = 0; kv < kvEnd; kv += 32) {
    __syncthreads();
    {  // cooperative V block (32 keys x 64 hd) -> LDS transposed
      const int kk = tid >> 3;   // key 0..31
      const int nn = tid & 7;    // hd group
      bh8 vv = *(const bh8*)(Vh + (size_t)(kv + kk) * HDn + nn * 8);
#pragma unroll
      for (int i = 0; i < 8; ++i) Vt[(nn * 8 + i) * 32 + kk] = vv[i];
      if (kv + 32 < kvEnd)
        __builtin_prefetch(Vh + (size_t)(kv + 32 + kk) * HDn + nn * 8, 0, 1);
    }
    __syncthreads();

    if (kv > qLast) continue;  // fully masked for this wave (wave-uniform)

    // ---- S = (Q K^T) for two 16-key tiles; K^T columns are contiguous in (B,H,T,HD)
    v8f s0 = {}, s1 = {};
#pragma unroll
    for (int c = 0; c < 2; ++c) {
      const bf16_t* kp0 = Kh + (size_t)(kv + nIdx) * HDn + c * 32 + 16 * lh;
      const bf16_t* kp1 = Kh + (size_t)(kv + 16 + nIdx) * HDn + c * 32 + 16 * lh;
      v16bf kb0 = cat8(*(const bh8*)kp0, *(const bh8*)(kp0 + 8));
      v16bf kb1 = cat8(*(const bh8*)kp1, *(const bh8*)(kp1 + 8));
      s0 = WMMA_BF16(qf[c], kb0, s0);
      s1 = WMMA_BF16(qf[c], kb1, s1);
    }

    // ---- online softmax (16-lane half = one set of 8 rows in C layout)
    const float scl = 0.125f;  // 1/sqrt(64)
    float p0[8], p1[8];
#pragma unroll
    for (int r = 0; r < 8; ++r) {
      const int qr = q0 + r + 8 * lh;
      float v0 = s0[r] * scl; if (kv + nIdx      > qr) v0 = -3.0e38f;
      float v1 = s1[r] * scl; if (kv + 16 + nIdx > qr) v1 = -3.0e38f;
      float rmax = fmaxf(v0, v1);
#pragma unroll
      for (int off = 8; off > 0; off >>= 1)
        rmax = fmaxf(rmax, __shfl_xor(rmax, off, 32));
      const float mn    = fmaxf(mRow[r], rmax);
      const float alpha = __expf(mRow[r] - mn);
      const float e0 = __expf(v0 - mn);
      const float e1 = __expf(v1 - mn);
      float rs = e0 + e1;
#pragma unroll
      for (int off = 8; off > 0; off >>= 1)
        rs += __shfl_xor(rs, off, 32);
      lRow[r] = lRow[r] * alpha + rs;
      mRow[r] = mn;
      p0[r] = e0; p1[r] = e1;
#pragma unroll
      for (int nt = 0; nt < 4; ++nt) oAcc[nt][r] *= alpha;
    }

    // ---- P (C layout) -> per-wave LDS -> re-fragment as WMMA A operand
    bf16_t* pw = Ps[wave];
#pragma unroll
    for (int r = 0; r < 8; ++r) {
      const int row = r + 8 * lh;
      pw[row * 32 + nIdx]      = (bf16_t)p0[r];
      pw[row * 32 + 16 + nIdx] = (bf16_t)p1[r];
    }
    asm volatile("s_wait_dscnt 0" ::: "memory");  // wave-local LDS RAW

    v16bf pf = frag_a_row32(pw, nIdx, lh);
#pragma unroll
    for (int nt = 0; nt < 4; ++nt) {
      v16bf vbf = frag_b_kmaj32(Vt + nt * 16 * 32, nIdx, lh);
      oAcc[nt] = WMMA_BF16(pf, vbf, oAcc[nt]);
    }
  }

  // ---- normalize + store attention output as bf16 in (B,T,D) for the out-proj GEMM
#pragma unroll
  for (int r = 0; r < 8; ++r) {
    const float inv = 1.0f / lRow[r];
    const int t = q0 + r + 8 * lh;
    bf16_t* op = Out + ((size_t)bb * Tn + t) * Dn + hh * HDn + nIdx;
#pragma unroll
    for (int nt = 0; nt < 4; ++nt)
      op[nt * 16] = (bf16_t)(oAcc[nt][r] * inv);
  }
}

// ---------------- host ----------------
extern "C" void kernel_launch(void* const* d_in, const int* in_sizes, int n_in,
                              void* d_out, int out_size, void* d_ws, size_t ws_size,
                              hipStream_t stream) {
  const float* x  = (const float*)d_in[0];
  const float* wq = (const float*)d_in[1];
  const float* bq = (const float*)d_in[2];
  const float* wk = (const float*)d_in[3];
  const float* bk = (const float*)d_in[4];
  const float* wv = (const float*)d_in[5];
  const float* bv = (const float*)d_in[6];
  const float* wo = (const float*)d_in[7];
  const float* bo = (const float*)d_in[8];

  const size_t XN = (size_t)Bn * Tn * Dn;  // 8,388,608
  const size_t WN = (size_t)Dn * Dn;       // 1,048,576

  // workspace layout (bf16 elements), total ~88 MB
  bf16_t* p   = (bf16_t*)d_ws;
  bf16_t* xb  = p; p += XN;
  bf16_t* wqb = p; p += WN;
  bf16_t* wkb = p; p += WN;
  bf16_t* wvb = p; p += WN;
  bf16_t* wob = p; p += WN;
  bf16_t* qb  = p; p += XN;   // (B,H,T,HD)
  bf16_t* kb  = p; p += XN;   // (B,H,T,HD)
  bf16_t* vb  = p; p += XN;   // (B,H,T,HD)
  bf16_t* ab  = p; p += XN;   // (B,T,D)

  auto cvt = [&](const float* src, bf16_t* dst, size_t n) {
    int n4 = (int)(n / 4);
    cvt_f32_bf16<<<(n4 + 255) / 256, 256, 0, stream>>>(src, dst, n4);
  };
  cvt(x, xb, XN);
  cvt(wq, wqb, WN);
  cvt(wk, wkb, WN);
  cvt(wv, wvb, WN);
  cvt(wo, wob, WN);

  dim3 gGrid(Dn / 64, (Bn * Tn) / 128);  // 16 x 64
  gemm_bf16_tile<<<gGrid, 256, 0, stream>>>(xb, wqb, bq, nullptr, qb, Dn, Dn, 1);
  gemm_bf16_tile<<<gGrid, 256, 0, stream>>>(xb, wkb, bk, nullptr, kb, Dn, Dn, 1);
  gemm_bf16_tile<<<gGrid, 256, 0, stream>>>(xb, wvb, bv, nullptr, vb, Dn, Dn, 1);

  dim3 fGrid(Tn / 128, Bn * Hn);         // 16 x 64
  flash_attn_bf16<<<fGrid, 256, 0, stream>>>(qb, kb, vb, ab);

  gemm_bf16_tile<<<gGrid, 256, 0, stream>>>(ab, wob, bo, (float*)d_out, nullptr, Dn, Dn, 0);
}
